// MultiheadAttention_65343632441694
// MI455X (gfx1250) — matrix operand order repaired and checked by reading, not compiled, not run on hardware
//
#include <hip/hip_runtime.h>
#include <hip/hip_bf16.h>

// ---------------------------------------------------------------------------
// MultiheadAttention forward for MI455X (gfx1250, wave32, WMMA)
//   B=2, S=2048, E=1024, H=16, DK=64
//   All GEMMs via v_wmma_f32_16x16x32_bf16 (bf16 in, f32 accumulate).
// ---------------------------------------------------------------------------

#define B_ 2
#define S_ 2048
#define E_ 1024
#define H_ 16
#define DK_ 64
#define M_ (B_ * S_)          // 4096 rows for the projection GEMMs

typedef __attribute__((ext_vector_type(16))) __bf16 v16bf;
typedef __attribute__((ext_vector_type(8)))  float  v8f;

union Frag {
    v16bf          v;
    unsigned short u[16];
    uint4          q[2];
};

__device__ __forceinline__ unsigned short f2bf(float f) {
    unsigned int u = __float_as_uint(f);
    u += 0x7FFFu + ((u >> 16) & 1u);      // round-to-nearest-even
    return (unsigned short)(u >> 16);
}

// A-matrix fragment (16x32, bf16, row-major source, leading dim `ld` halves).
// CDNA5 layout: lane m=lane&15, element j -> K = (j>>3)*16 + hi*8 + (j&7)
// => two contiguous 8-half (16-byte) runs at k+hi*8 and k+16+hi*8.
__device__ __forceinline__ v16bf load_a_frag(const unsigned short* __restrict__ base,
                                             int ld, int row, int k, int hi) {
    Frag f;
    const unsigned short* p = base + (size_t)row * ld + k + hi * 8;
    f.q[0] = *(const uint4*)p;
    f.q[1] = *(const uint4*)(p + 16);
    return f.v;
}

// B-matrix fragment (32x16, bf16, column-major: lane holds column `col`).
// CDNA5 layout: lane n=lane&15, element j -> K = (j>>3)*16 + hi*8 + (j&7)
// => two contiguous 8-half (16-byte) runs at k+hi*8 and k+16+hi*8.
__device__ __forceinline__ v16bf load_b_frag(const unsigned short* __restrict__ base,
                                             int ld, int col, int k, int hi) {
    Frag f;
    const unsigned short* p = base + (size_t)col * ld + k + hi * 8;
    f.q[0] = *(const uint4*)p;
    f.q[1] = *(const uint4*)(p + 16);
    return f.v;
}

// ---------------------------------------------------------------------------
// fp32 -> bf16 conversion (grid-stride)
// ---------------------------------------------------------------------------
__global__ void cvt_f32_to_bf16(const float* __restrict__ src,
                                unsigned short* __restrict__ dst, int n) {
    int i = blockIdx.x * blockDim.x + threadIdx.x;
    int stride = gridDim.x * blockDim.x;
    for (; i < n; i += stride) dst[i] = f2bf(src[i]);
}

// ---------------------------------------------------------------------------
// GEMM: out = A[4096,1024] @ W^T[1024,1024] + bias, via WMMA bf16.
//   mode 0: store bf16 head-split      [B,H,S,DK]
//   mode 1: store bf16 head-split-T    [B,H,DK,S]   (for V)
//   mode 2: store f32 row-major        [4096,1024]  (final output)
// Block: 256 threads = 8 waves in 4x2 -> 256x128 block tile, 64x64 per wave.
// ---------------------------------------------------------------------------
__global__ __launch_bounds__(256)
void gemm_bf16(const unsigned short* __restrict__ A,    // [4096,1024] bf16
               const unsigned short* __restrict__ W,    // [1024,1024] bf16
               const float* __restrict__ bias,          // [1024]
               unsigned short* __restrict__ outBf,
               float* __restrict__ outF,
               int mode) {
    const int lane = threadIdx.x & 31;
    const int hi   = lane >> 4;
    const int l15  = lane & 15;
    const int wave = threadIdx.x >> 5;
    const int m0   = blockIdx.y * 256 + (wave >> 1) * 64;
    const int n0   = blockIdx.x * 128 + (wave & 1) * 64;

    v8f acc[4][4];
#pragma unroll
    for (int mi = 0; mi < 4; ++mi)
#pragma unroll
        for (int ni = 0; ni < 4; ++ni)
            acc[mi][ni] = (v8f){0.f, 0.f, 0.f, 0.f, 0.f, 0.f, 0.f, 0.f};

    for (int k = 0; k < E_; k += 32) {
        if (k + 32 < E_) {   // -> global_prefetch_b8
            __builtin_prefetch(A + (size_t)(m0 + l15) * E_ + k + 32, 0, 0);
            __builtin_prefetch(W + (size_t)(n0 + l15) * E_ + k + 32, 0, 0);
        }
        v16bf a[4], b[4];
#pragma unroll
        for (int mi = 0; mi < 4; ++mi)
            a[mi] = load_a_frag(A, E_, m0 + mi * 16 + l15, k, hi);
#pragma unroll
        for (int ni = 0; ni < 4; ++ni)
            b[ni] = load_b_frag(W, E_, n0 + ni * 16 + l15, k, hi);
#pragma unroll
        for (int mi = 0; mi < 4; ++mi)
#pragma unroll
            for (int ni = 0; ni < 4; ++ni)
                acc[mi][ni] = __builtin_amdgcn_wmma_f32_16x16x32_bf16(
                    false, a[mi], false, b[ni], (short)0, acc[mi][ni], false, false);
    }

    // Epilogue: C layout -> element r maps to (M = m0+mi*16+hi*8+r, N = n0+ni*16+l15)
#pragma unroll
    for (int ni = 0; ni < 4; ++ni) {
        const int n  = n0 + ni * 16 + l15;
        const float bv = bias[n];
#pragma unroll
        for (int mi = 0; mi < 4; ++mi) {
#pragma unroll
            for (int r = 0; r < 8; ++r) {
                const int m = m0 + mi * 16 + hi * 8 + r;
                const float val = acc[mi][ni][r] + bv;
                if (mode == 2) {
                    outF[(size_t)m * E_ + n] = val;
                } else {
                    const int bb = m >> 11, s = m & (S_ - 1);
                    const int h  = n >> 6,  d = n & (DK_ - 1);
                    if (mode == 0)
                        outBf[(((size_t)bb * H_ + h) * S_ + s) * DK_ + d] = f2bf(val);
                    else
                        outBf[(((size_t)bb * H_ + h) * DK_ + d) * S_ + s] = f2bf(val);
                }
            }
        }
    }
}

// ---------------------------------------------------------------------------
// Flash attention: per (b,h), streams 32-key blocks with online softmax.
// Block = 128 threads = 4 waves; each wave owns 16 query rows.
// Q,K: [B,H,S,64] bf16; Vt: [B,H,64,S] bf16; mask: [B,S] int; Xa: [B,S,E] bf16
// ---------------------------------------------------------------------------
__global__ __launch_bounds__(128)
void attn_flash(const unsigned short* __restrict__ Qh,
                const unsigned short* __restrict__ Kh,
                const unsigned short* __restrict__ Vt,
                const int* __restrict__ mask,
                unsigned short* __restrict__ Xa) {
    __shared__ __align__(16) unsigned short pLds[4][16][32];  // per-wave P tile

    const int lane = threadIdx.x & 31;
    const int hi   = lane >> 4;
    const int l15  = lane & 15;
    const int wave = threadIdx.x >> 5;
    const int b = blockIdx.z, h = blockIdx.y;
    const int q0 = blockIdx.x * 64 + wave * 16;

    const unsigned short* Qp = Qh + ((size_t)(b * H_ + h) * S_) * DK_;
    const unsigned short* Kp = Kh + ((size_t)(b * H_ + h) * S_) * DK_;
    const unsigned short* Vp = Vt + ((size_t)(b * H_ + h) * DK_) * S_;
    const int* mp = mask + (size_t)b * S_;

    // Q fragments for this wave's 16 rows (kept in registers for all keys)
    v16bf aq0 = load_a_frag(Qp, DK_, q0 + l15, 0,  hi);
    v16bf aq1 = load_a_frag(Qp, DK_, q0 + l15, 32, hi);

    v8f o[4];
#pragma unroll
    for (int f = 0; f < 4; ++f) o[f] = (v8f){0.f, 0.f, 0.f, 0.f, 0.f, 0.f, 0.f, 0.f};
    float mrow[8], lrow[8];
#pragma unroll
    for (int r = 0; r < 8; ++r) { mrow[r] = -3.0e38f; lrow[r] = 0.f; }

    for (int kb = 0; kb < S_; kb += 32) {
        // ---- scores = (Q K^T) / sqrt(DK), two 16-key tiles ----
        v8f sc[2];
#pragma unroll
        for (int t = 0; t < 2; ++t) {
            const int key = kb + t * 16 + l15;
            v16bf bk0 = load_b_frag(Kp, DK_, key, 0,  hi);
            v16bf bk1 = load_b_frag(Kp, DK_, key, 32, hi);
            v8f z = (v8f){0.f, 0.f, 0.f, 0.f, 0.f, 0.f, 0.f, 0.f};
            z = __builtin_amdgcn_wmma_f32_16x16x32_bf16(false, aq0, false, bk0, (short)0, z, false, false);
            z = __builtin_amdgcn_wmma_f32_16x16x32_bf16(false, aq1, false, bk1, (short)0, z, false, false);
            sc[t] = z;
        }
        // scale + key mask (mask < 1 -> -1e9, applied after scaling like reference)
#pragma unroll
        for (int t = 0; t < 2; ++t) {
            const int key = kb + t * 16 + l15;
            const bool dead = (mp[key] < 1);
#pragma unroll
            for (int r = 0; r < 8; ++r) {
                float v = sc[t][r] * 0.125f;       // 1/sqrt(64)
                sc[t][r] = dead ? -1.0e9f : v;
            }
        }
        // ---- online softmax: row max (reduce over 16 lanes of each half) ----
        float alpha[8];
#pragma unroll
        for (int r = 0; r < 8; ++r) {
            float tmax = fmaxf(sc[0][r], sc[1][r]);
#pragma unroll
            for (int xm = 1; xm <= 8; xm <<= 1)
                tmax = fmaxf(tmax, __shfl_xor(tmax, xm, 32));
            const float mnew = fmaxf(mrow[r], tmax);
            alpha[r] = __expf(mrow[r] - mnew);
            mrow[r]  = mnew;
        }
        // ---- P = exp(scores - m), accumulate row sums, stage P to LDS ----
        float ladd[8];
#pragma unroll
        for (int r = 0; r < 8; ++r) ladd[r] = 0.f;
#pragma unroll
        for (int t = 0; t < 2; ++t)
#pragma unroll
            for (int r = 0; r < 8; ++r) {
                const float p = __expf(sc[t][r] - mrow[r]);
                ladd[r] += p;
                pLds[wave][hi * 8 + r][t * 16 + l15] = f2bf(p);
            }
#pragma unroll
        for (int r = 0; r < 8; ++r) {
            float s = ladd[r];
#pragma unroll
            for (int xm = 1; xm <= 8; xm <<= 1)
                s += __shfl_xor(s, xm, 32);
            lrow[r] = lrow[r] * alpha[r] + s;
        }
        // rescale running output
#pragma unroll
        for (int f = 0; f < 4; ++f)
#pragma unroll
            for (int r = 0; r < 8; ++r) o[f][r] *= alpha[r];

        // ---- read P back as A-fragment (16x32 over this key block) ----
        // (same-wave LDS ordering: DS ops are in-order; compiler inserts dscnt waits)
        Frag pf;
        const unsigned short* pp = &pLds[wave][l15][hi * 8];
        pf.q[0] = *(const uint4*)pp;
        pf.q[1] = *(const uint4*)(pp + 16);

        // ---- O += P @ V : V^T stored [DK,S] so B-frag columns are contiguous ----
#pragma unroll
        for (int dt = 0; dt < 4; ++dt) {
            v16bf bv = load_b_frag(Vp, S_, dt * 16 + l15, kb, hi);
            o[dt] = __builtin_amdgcn_wmma_f32_16x16x32_bf16(
                false, pf.v, false, bv, (short)0, o[dt], false, false);
        }
    }

    // ---- finalize: O /= l, write bf16 [B,S,E] for the output projection ----
#pragma unroll
    for (int r = 0; r < 8; ++r) lrow[r] = 1.0f / lrow[r];
#pragma unroll
    for (int dt = 0; dt < 4; ++dt)
#pragma unroll
        for (int r = 0; r < 8; ++r) {
            const int s   = q0 + hi * 8 + r;
            const int col = h * DK_ + dt * 16 + l15;
            Xa[((size_t)b * S_ + s) * E_ + col] = f2bf(o[dt][r] * lrow[r]);
        }
}

// ---------------------------------------------------------------------------
// Launch
// ---------------------------------------------------------------------------
extern "C" void kernel_launch(void* const* d_in, const int* in_sizes, int n_in,
                              void* d_out, int out_size, void* d_ws, size_t ws_size,
                              hipStream_t stream) {
    const float* query  = (const float*)d_in[0];
    const float* key_in = (const float*)d_in[1];
    const float* value  = (const float*)d_in[2];
    const int*   mask   = (const int*)  d_in[3];
    const float* Wq = (const float*)d_in[4];  const float* bq = (const float*)d_in[5];
    const float* Wk = (const float*)d_in[6];  const float* bk = (const float*)d_in[7];
    const float* Wv = (const float*)d_in[8];  const float* bv = (const float*)d_in[9];
    const float* Wo = (const float*)d_in[10]; const float* bo = (const float*)d_in[11];
    float* out = (float*)d_out;

    const int ACT = M_ * E_;   // 4096*1024
    const int WSZ = E_ * E_;   // 1024*1024

    char* ws = (char*)d_ws;
    size_t off = 0;
    auto carve = [&](size_t bytes) {
        void* p = ws + off;
        off += (bytes + 255) & ~(size_t)255;
        return p;
    };
    unsigned short* qbf  = (unsigned short*)carve((size_t)ACT * 2);
    unsigned short* kbf  = (unsigned short*)carve((size_t)ACT * 2);
    unsigned short* vbf  = (unsigned short*)carve((size_t)ACT * 2);
    unsigned short* wqbf = (unsigned short*)carve((size_t)WSZ * 2);
    unsigned short* wkbf = (unsigned short*)carve((size_t)WSZ * 2);
    unsigned short* wvbf = (unsigned short*)carve((size_t)WSZ * 2);
    unsigned short* wobf = (unsigned short*)carve((size_t)WSZ * 2);
    unsigned short* Qh   = (unsigned short*)carve((size_t)ACT * 2);  // [B,H,S,DK]
    unsigned short* Kh   = (unsigned short*)carve((size_t)ACT * 2);  // [B,H,S,DK]
    unsigned short* Vt   = (unsigned short*)carve((size_t)ACT * 2);  // [B,H,DK,S]
    unsigned short* Xa   = (unsigned short*)carve((size_t)ACT * 2);  // [B,S,E]

    // 1) fp32 -> bf16 conversions
    cvt_f32_to_bf16<<<2048, 256, 0, stream>>>(query,  qbf,  ACT);
    cvt_f32_to_bf16<<<2048, 256, 0, stream>>>(key_in, kbf,  ACT);
    cvt_f32_to_bf16<<<2048, 256, 0, stream>>>(value,  vbf,  ACT);
    cvt_f32_to_bf16<<<1024, 256, 0, stream>>>(Wq, wqbf, WSZ);
    cvt_f32_to_bf16<<<1024, 256, 0, stream>>>(Wk, wkbf, WSZ);
    cvt_f32_to_bf16<<<1024, 256, 0, stream>>>(Wv, wvbf, WSZ);
    cvt_f32_to_bf16<<<1024, 256, 0, stream>>>(Wo, wobf, WSZ);

    // 2) projections (WMMA): grid = (N/128, M/256)
    dim3 ggrid(E_ / 128, M_ / 256);
    gemm_bf16<<<ggrid, 256, 0, stream>>>(qbf, wqbf, bq, Qh, nullptr, 0);
    gemm_bf16<<<ggrid, 256, 0, stream>>>(kbf, wkbf, bk, Kh, nullptr, 0);
    gemm_bf16<<<ggrid, 256, 0, stream>>>(vbf, wvbf, bv, Vt, nullptr, 1);

    // 3) attention (WMMA flash): grid = (S/64, H, B)
    attn_flash<<<dim3(S_ / 64, H_, B_), 128, 0, stream>>>(Qh, Kh, Vt, mask, Xa);

    // 4) output projection -> f32 d_out
    gemm_bf16<<<ggrid, 256, 0, stream>>>(Xa, wobf, bo, nullptr, out, 2);
}
